// SDG_LLM_18253611008483
// MI455X (gfx1250) — matrix operand order repaired and checked by reading, not compiled, not run on hardware
//
#include <hip/hip_runtime.h>
#include <hip/hip_bf16.h>

typedef __attribute__((ext_vector_type(2)))  float  v2f;
typedef __attribute__((ext_vector_type(8)))  float  v8f;
typedef __attribute__((ext_vector_type(8)))  __bf16 v8bf;
typedef __attribute__((ext_vector_type(16))) __bf16 v16bf;

#define N_NODES 128
#define P_DIM   128
#define H_DIM   4096
#define S_DIM   128
#define SH      (S_DIM * H_DIM)   /* 524288 columns of the big GEMM */

// ---------------------------------------------------------------------------
// Kernel 1: p_hat = p @ W_proj^T + b   (exact fp32 via V_WMMA_F32_16X16X4_F32)
// A = p [128 x 128], B[k][h] = W_proj[h][k]  (contiguous b64 loads per lane)
// ---------------------------------------------------------------------------
__global__ __launch_bounds__(256)
void phat_kernel(const float* __restrict__ p, const float* __restrict__ W,
                 const float* __restrict__ bias, float* __restrict__ phat) {
  const int lane = threadIdx.x & 31;
  const int wave = threadIdx.x >> 5;
  const int tile = blockIdx.x * 8 + wave;   // 2048 tiles total
  const int cT = tile >> 3;                 // 0..255  (H tiles)
  const int mT = tile & 7;                  // 0..7    (N tiles)
  const int l15  = lane & 15;
  const int half = lane >> 4;

  const float* aRow = p + (mT * 16 + l15) * P_DIM;
  const float* bRow = W + (cT * 16 + l15) * P_DIM;

  v8f acc = {};
  for (int k = 0; k < P_DIM; k += 4) {
    v2f a = *(const v2f*)(aRow + k + half * 2);   // A: K=(half*2)+{0,1}
    v2f b = *(const v2f*)(bRow + k + half * 2);   // B: rows (half*2)+{0,1}, col l15
    acc = __builtin_amdgcn_wmma_f32_16x16x4_f32(false, a, false, b,
                                                (short)0, acc, false, false);
  }
  const float bv = bias[cT * 16 + l15];
  for (int v = 0; v < 8; ++v) {
    const int row = mT * 16 + v + half * 8;       // C/D layout: VGPR v -> M=v / v+8
    phat[(size_t)row * H_DIM + cT * 16 + l15] = acc[v] + bv;
  }
}

// ---------------------------------------------------------------------------
// Kernel 2: inv_norm[n] = 1 / max(||p_hat[n,:]||, 1e-12)
// ---------------------------------------------------------------------------
__global__ __launch_bounds__(256)
void norm_kernel(const float* __restrict__ phat, float* __restrict__ invn) {
  __shared__ float red[256];
  const int n = blockIdx.x;
  const float* row = phat + (size_t)n * H_DIM;
  float s = 0.f;
  for (int i = threadIdx.x; i < H_DIM; i += 256) { float x = row[i]; s += x * x; }
  red[threadIdx.x] = s;
  __syncthreads();
  for (int off = 128; off > 0; off >>= 1) {
    if (threadIdx.x < off) red[threadIdx.x] += red[threadIdx.x + off];
    __syncthreads();
  }
  if (threadIdx.x == 0) invn[n] = 1.0f / fmaxf(sqrtf(red[0]), 1e-12f);
}

// ---------------------------------------------------------------------------
// Kernel 3: sim[m,n] = (p_hat[m,:] . p_hat[n,:]) * invn[m] * invn[n]
// 64 tiles, 4 waves/tile each covering K/4=1024, LDS partial-sum reduce.
// ---------------------------------------------------------------------------
__global__ __launch_bounds__(128)
void gram_kernel(const float* __restrict__ phat, const float* __restrict__ invn,
                 float* __restrict__ sim) {
  __shared__ float partial[4][8][32];
  const int lane = threadIdx.x & 31;
  const int wave = threadIdx.x >> 5;   // K-chunk 0..3
  const int tile = blockIdx.x;         // 0..63
  const int mT = tile >> 3, nT = tile & 7;
  const int l15 = lane & 15, half = lane >> 4;

  const float* aRow = phat + (size_t)(mT * 16 + l15) * H_DIM;
  const float* bRow = phat + (size_t)(nT * 16 + l15) * H_DIM;   // B[k][n]=phat[n][k]

  v8f acc = {};
  const int k0 = wave * (H_DIM / 4);
  for (int k = k0; k < k0 + H_DIM / 4; k += 4) {
    v2f a = *(const v2f*)(aRow + k + half * 2);
    v2f b = *(const v2f*)(bRow + k + half * 2);
    acc = __builtin_amdgcn_wmma_f32_16x16x4_f32(false, a, false, b,
                                                (short)0, acc, false, false);
  }
  for (int v = 0; v < 8; ++v) partial[wave][v][lane] = acc[v];
  __syncthreads();
  if (wave == 0) {
    const float invN = invn[nT * 16 + l15];
    for (int v = 0; v < 8; ++v) {
      float s = partial[0][v][lane] + partial[1][v][lane] +
                partial[2][v][lane] + partial[3][v][lane];
      const int row = mT * 16 + v + half * 8;
      sim[row * N_NODES + nT * 16 + l15] = s * invn[row] * invN;
    }
  }
}

// ---------------------------------------------------------------------------
// Kernel 4: per row m: rs = sum_n sim[m,n]; rs += (rs==0);
//           a = sim/rs -> split into bf16 hi + bf16 lo (Markidis split).
// ---------------------------------------------------------------------------
__global__ __launch_bounds__(128)
void prep_kernel(const float* __restrict__ sim, __bf16* __restrict__ ahi,
                 __bf16* __restrict__ alo) {
  __shared__ float red[128];
  const int m = blockIdx.x;
  const float v = sim[m * N_NODES + threadIdx.x];
  red[threadIdx.x] = v;
  __syncthreads();
  for (int off = 64; off > 0; off >>= 1) {
    if (threadIdx.x < off) red[threadIdx.x] += red[threadIdx.x + off];
    __syncthreads();
  }
  float rs = red[0];
  rs = rs + (rs == 0.0f ? 1.0f : 0.0f);   // reference's div-by-zero guard
  const float a = v / rs;
  const __bf16 h = (__bf16)a;
  ahi[m * N_NODES + threadIdx.x] = h;
  alo[m * N_NODES + threadIdx.x] = (__bf16)(a - (float)h);
}

// ---------------------------------------------------------------------------
// Kernel 5 (dominant): t'[m,c] = sum_n A[m,n] * T[n,c],  T = t as [128 x 524288].
// bf16 hi/lo split, 3x v_wmma_f32_16x16x32_bf16 per (m-tile, k-block).
// Block = 8 waves covers a 128-column strip over ALL 128 rows -> t read once.
// A (hi+lo, 64 KB) staged in LDS once per block.
// ---------------------------------------------------------------------------
__global__ __launch_bounds__(256)
void sattn_kernel(const float* __restrict__ t, const __bf16* __restrict__ Ahi,
                  const __bf16* __restrict__ Alo, float* __restrict__ outp) {
  __shared__ __bf16 sAhi[N_NODES * N_NODES];
  __shared__ __bf16 sAlo[N_NODES * N_NODES];
  {
    const uint4* g0 = (const uint4*)Ahi;
    const uint4* g1 = (const uint4*)Alo;
    uint4* s0 = (uint4*)sAhi;
    uint4* s1 = (uint4*)sAlo;
    for (int i = threadIdx.x; i < (N_NODES * N_NODES) / 8; i += 256) {
      s0[i] = g0[i];
      s1[i] = g1[i];
    }
  }
  __syncthreads();

  const int lane = threadIdx.x & 31;
  const int wave = threadIdx.x >> 5;
  const int l15  = lane & 15;
  const int half = lane >> 4;
  const size_t c = (size_t)blockIdx.x * 128 + wave * 16 + l15;

  v8f acc[8];
  for (int mt = 0; mt < 8; ++mt) acc[mt] = (v8f){};

  for (int kb = 0; kb < 4; ++kb) {               // K = 128 in blocks of 32
    const int kBase = kb * 32;
    // B fragment (32x16 bf16): element j = T[kBase + half*16 + j][c]
    v16bf bhi, blo;
    const float* tc = t + (size_t)(kBase + half * 16) * SH + c;
    for (int j = 0; j < 16; ++j) {
      const float x = tc[(size_t)j * SH];
      const __bf16 h = (__bf16)x;
      bhi[j] = h;
      blo[j] = (__bf16)(x - (float)h);
    }
    for (int mt = 0; mt < 8; ++mt) {
      // A fragment (16x32 bf16): lane row = mt*16+l15,
      // elems 0..15 -> K = kBase + half*16 + j
      const int o1 = (mt * 16 + l15) * N_NODES + kBase + half * 16;
      const int o2 = o1 + 8;
      v8bf h1 = *(const v8bf*)(sAhi + o1);
      v8bf h2 = *(const v8bf*)(sAhi + o2);
      v8bf l1 = *(const v8bf*)(sAlo + o1);
      v8bf l2 = *(const v8bf*)(sAlo + o2);
      v16bf ahiF = __builtin_shufflevector(h1, h2, 0,1,2,3,4,5,6,7,8,9,10,11,12,13,14,15);
      v16bf aloF = __builtin_shufflevector(l1, l2, 0,1,2,3,4,5,6,7,8,9,10,11,12,13,14,15);
      acc[mt] = __builtin_amdgcn_wmma_f32_16x16x32_bf16(false, ahiF, false, bhi,
                                                        (short)0, acc[mt], false, false);
      acc[mt] = __builtin_amdgcn_wmma_f32_16x16x32_bf16(false, ahiF, false, blo,
                                                        (short)0, acc[mt], false, false);
      acc[mt] = __builtin_amdgcn_wmma_f32_16x16x32_bf16(false, aloF, false, bhi,
                                                        (short)0, acc[mt], false, false);
    }
  }
  for (int mt = 0; mt < 8; ++mt)
    for (int v = 0; v < 8; ++v) {
      const int row = mt * 16 + v + half * 8;
      outp[(size_t)row * SH + c] = acc[mt][v];
    }
}

// ---------------------------------------------------------------------------
extern "C" void kernel_launch(void* const* d_in, const int* in_sizes, int n_in,
                              void* d_out, int out_size, void* d_ws, size_t ws_size,
                              hipStream_t stream) {
  const float* t = (const float*)d_in[0];   // [128,128,4096]
  const float* p = (const float*)d_in[1];   // [128,128]
  const float* W = (const float*)d_in[2];   // [4096,128]
  const float* b = (const float*)d_in[3];   // [4096]

  float* out    = (float*)d_out;
  float* tprime = out;                            // 128*128*4096 elements
  float* phat   = out + (size_t)N_NODES * SH;     // then 128*4096 elements

  char*   ws   = (char*)d_ws;
  float*  sim  = (float*)(ws);            // 64 KB  : cosine sim matrix
  __bf16* ahi  = (__bf16*)(ws + 65536);   // 32 KB  : scaled sim, bf16 hi
  __bf16* alo  = (__bf16*)(ws + 98304);   // 32 KB  : scaled sim, bf16 lo
  float*  invn = (float*)(ws + 131072);   // 512 B  : inverse row norms

  phat_kernel <<<256, 256, 0, stream>>>(p, W, b, phat);
  norm_kernel <<<128, 256, 0, stream>>>(phat, invn);
  gram_kernel <<< 64, 128, 0, stream>>>(phat, invn, sim);
  prep_kernel <<<128, 128, 0, stream>>>(sim, ahi, alo);
  sattn_kernel<<<SH / 128, 256, 0, stream>>>(t, ahi, alo, tprime);
}
